// MotifConv_25383256719489
// MI455X (gfx1250) — matrix operand order repaired and checked by reading, not compiled, not run on hardware
//
#include <hip/hip_runtime.h>

#define N_NODES 50000
#define N_EDGES 800000
#define FDIM 128
#define DDIM 64
#define OUTC (13 * DDIM)   // 832

typedef __attribute__((ext_vector_type(16))) __bf16 v16bf;
typedef __attribute__((ext_vector_type(8)))  __bf16 v8bf;
typedef __attribute__((ext_vector_type(8)))  float  v8f;

// ---------------------------------------------------------------------------
// Workspace layout (bytes from d_ws):
//   rsts_bf16 : 14 * N * 128 * 2  = 179.2 MB   (rsts[0] = h, rsts[1..13] = motif props)
//   agg_f32   :      N * 128 * 4  =  25.6 MB   (edge-set-0 fp32 accumulator)
//   weight_pk : 128*128 bf16 (WMMA-B packed)
//   root_pk   : 128*128 bf16
//   wa_pk     : 128*64  bf16
//   motif_pk  : 13*1664*64 bf16
// ---------------------------------------------------------------------------

// ---------------- zero fill ----------------
__global__ void zero_kernel(uint4* p, long long n16) {
    long long i = (long long)blockIdx.x * blockDim.x + threadIdx.x;
    if (i < n16) { uint4 z; z.x = z.y = z.z = z.w = 0u; p[i] = z; }
}

// ------------- pack fp32 weights into WMMA B-fragment bf16 layout ----------
// Fragment layout per 32(K) x 16(N) tile, 32 lanes x 16 halves:
//   lane<16 : col = lane,   K = kt*32 + (h<8 ? h : 16 + h-8)
//   lane>=16: col = lane-16, K = kt*32 + 8 + (h<8 ? h : 16 + h-8)
__device__ __forceinline__ void pack_one(const float* __restrict__ src,
                                         __bf16* __restrict__ dst,
                                         int idx, int ncols) {
    int h    = idx & 15;
    int lane = (idx >> 4) & 31;
    int rem  = idx >> 9;
    int NT   = ncols >> 4;
    int nt   = rem % NT;
    int kt   = rem / NT;
    int k    = kt * 32 + ((h & 8) << 1) + ((lane & 16) >> 1) + (h & 7);
    int n    = nt * 16 + (lane & 15);
    dst[idx] = (__bf16)src[(size_t)k * ncols + n];
}

__global__ void pack_kernel(const float* __restrict__ weight,
                            const float* __restrict__ root,
                            const float* __restrict__ wa,
                            const float* __restrict__ motif_w,
                            __bf16* __restrict__ wpk, __bf16* __restrict__ rpk,
                            __bf16* __restrict__ wapk, __bf16* __restrict__ mpk) {
    int idx = blockIdx.x * blockDim.x + threadIdx.x;
    const int SZ_W = 128 * 128, SZ_WA = 128 * 64, SZ_M = 13 * FDIM * DDIM; // 1664*64
    if (idx < SZ_W) { pack_one(weight, wpk, idx, 128); return; }
    idx -= SZ_W;
    if (idx < SZ_W) { pack_one(root, rpk, idx, 128); return; }
    idx -= SZ_W;
    if (idx < SZ_WA) { pack_one(wa, wapk, idx, 64); return; }
    idx -= SZ_WA;
    if (idx < 13 * SZ_M) {
        int g = idx / SZ_M, li = idx % SZ_M;
        pack_one(motif_w + (size_t)g * SZ_M, mpk + (size_t)g * SZ_M, li, 64);
    }
}

// ---------------- edge-set-0 propagate (fp32, full precision) ----------------
__global__ void prop0_kernel(const float* __restrict__ x,
                             const long long* __restrict__ ei,
                             const float* __restrict__ ew,
                             float* __restrict__ agg) {
    long long id = (long long)blockIdx.x * blockDim.x + threadIdx.x; // E * 32
    int q = (int)(id & 31);
    long long e = id >> 5;
    if (e >= N_EDGES) return;
    long long s = ei[e];            // [0][0][e]
    long long d = ei[N_EDGES + e];  // [0][1][e]
    float w = ew[e];
    float4 v = *((const float4*)(x + (size_t)s * FDIM) + q);
    float* ap = agg + (size_t)d * FDIM + q * 4;
    atomicAdd(ap + 0, w * v.x);
    atomicAdd(ap + 1, w * v.y);
    atomicAdd(ap + 2, w * v.z);
    atomicAdd(ap + 3, w * v.w);
}

// ---------------- WMMA A-fragment loaders (16-bit 16x32 ISA layout) ----------
__device__ __forceinline__ v16bf load_a_f32(const float* __restrict__ A,
                                            int m0, int lane, int k0) {
    int row = m0 + (lane & 15);
    if (row >= N_NODES) row = N_NODES - 1;           // masked at store time
    int kb = k0 + ((lane >> 4) << 3);                // +8 for upper half-wave
    const float* rp = A + (size_t)row * FDIM + kb;
    float4 p0 = *(const float4*)(rp);
    float4 p1 = *(const float4*)(rp + 4);
    float4 p2 = *(const float4*)(rp + 16);
    float4 p3 = *(const float4*)(rp + 20);
    v16bf a;
    a[0]  = (__bf16)p0.x; a[1]  = (__bf16)p0.y; a[2]  = (__bf16)p0.z; a[3]  = (__bf16)p0.w;
    a[4]  = (__bf16)p1.x; a[5]  = (__bf16)p1.y; a[6]  = (__bf16)p1.z; a[7]  = (__bf16)p1.w;
    a[8]  = (__bf16)p2.x; a[9]  = (__bf16)p2.y; a[10] = (__bf16)p2.z; a[11] = (__bf16)p2.w;
    a[12] = (__bf16)p3.x; a[13] = (__bf16)p3.y; a[14] = (__bf16)p3.z; a[15] = (__bf16)p3.w;
    return a;
}

__device__ __forceinline__ v16bf load_a_bf16(const __bf16* __restrict__ A,
                                             int m0, int lane, int k0) {
    int row = m0 + (lane & 15);
    if (row >= N_NODES) row = N_NODES - 1;
    int kb = k0 + ((lane >> 4) << 3);
    const __bf16* rp = A + (size_t)row * FDIM + kb;
    v8bf lo = *(const v8bf*)(rp);
    v8bf hi = *(const v8bf*)(rp + 16);
    v16bf a;
#pragma unroll
    for (int t = 0; t < 8; ++t) { a[t] = lo[t]; a[t + 8] = hi[t]; }
    return a;
}

// ---------------- h = agg@weight + x@root + bias  ->  rsts[0] (bf16) --------
__global__ __launch_bounds__(256)
void h_gemm_kernel(const float* __restrict__ agg, const float* __restrict__ x,
                   const float* __restrict__ bias,
                   const __bf16* __restrict__ wpk, const __bf16* __restrict__ rpk,
                   __bf16* __restrict__ h_out) {
    int lane = threadIdx.x & 31;
    int wave = threadIdx.x >> 5;
    int m0 = blockIdx.x * 128 + wave * 16;

    v8f acc[8];
#pragma unroll
    for (int nt = 0; nt < 8; ++nt)
#pragma unroll
        for (int t = 0; t < 8; ++t) acc[nt][t] = 0.f;

#pragma unroll
    for (int srcSel = 0; srcSel < 2; ++srcSel) {
        const float* A = srcSel ? x : agg;
        const __bf16* B = srcSel ? rpk : wpk;
#pragma unroll
        for (int s = 0; s < 4; ++s) {
            v16bf a = load_a_f32(A, m0, lane, s * 32);
#pragma unroll
            for (int nt = 0; nt < 8; ++nt) {
                v16bf b = *(const v16bf*)(B + ((size_t)((s * 8 + nt) * 32 + lane) << 4));
                acc[nt] = __builtin_amdgcn_wmma_f32_16x16x32_bf16(
                    false, a, false, b, (short)0, acc[nt], false, false);
            }
        }
    }

    int row_lo = m0 + ((lane >> 4) << 3);  // upper half-wave owns rows +8..15
    int col0 = lane & 15;
#pragma unroll
    for (int nt = 0; nt < 8; ++nt) {
        int col = nt * 16 + col0;
        float bb = bias[col];
#pragma unroll
        for (int r = 0; r < 8; ++r) {
            int row = row_lo + r;
            if (row < N_NODES)
                h_out[(size_t)row * FDIM + col] = (__bf16)(acc[nt][r] + bb);
        }
    }
}

// ---------------- motif propagates: bf16 gather + packed-bf16 atomic scatter -
__global__ void motif_prop_kernel(const long long* __restrict__ ei,
                                  const float* __restrict__ ew,
                                  const __bf16* __restrict__ h0,
                                  __bf16* __restrict__ rsts) {
    long long id = (long long)blockIdx.x * blockDim.x + threadIdx.x;
    int p = (int)(id & 63);                  // bf16 pair index (0..63)
    long long rest = id >> 6;
    long long e = rest % N_EDGES;
    int g = 1 + (int)(rest / N_EDGES);       // graph 1..13
    if (g > 13) return;
    long long s = ei[((long long)g * 2 + 0) * N_EDGES + e];
    long long d = ei[((long long)g * 2 + 1) * N_EDGES + e];
    float w = ew[(long long)g * N_EDGES + e];

    unsigned int pk = *(const unsigned int*)(h0 + (size_t)s * FDIM + p * 2);
    float f0 = __uint_as_float(pk << 16);
    float f1 = __uint_as_float(pk & 0xFFFF0000u);
    __bf16 r0 = (__bf16)(f0 * w);
    __bf16 r1 = (__bf16)(f1 * w);
    unsigned int data = (unsigned int)__builtin_bit_cast(unsigned short, r0) |
                        ((unsigned int)__builtin_bit_cast(unsigned short, r1) << 16);
    unsigned long long addr =
        (unsigned long long)(rsts + (size_t)g * N_NODES * FDIM + (size_t)d * FDIM + p * 2);
    // CDNA5 packed bf16 atomic add, no return (STOREcnt path)
    asm volatile("global_atomic_pk_add_bf16 %0, %1, off" :: "v"(addr), "v"(data) : "memory");
}

// ---------------- fused motif GEMM + attention + output ---------------------
__global__ __launch_bounds__(256)
void motif_gemm_kernel(const __bf16* __restrict__ rsts,
                       const __bf16* __restrict__ wapk,
                       const __bf16* __restrict__ mpk,
                       const float* __restrict__ ba,
                       const float* __restrict__ motif_b,
                       float* __restrict__ out) {
    int lane = threadIdx.x & 31;
    int wave = threadIdx.x >> 5;
    int m0 = blockIdx.x * 128 + wave * 16;
    int g = blockIdx.y;      // 0..12
    int i = g + 1;           // motif graph index in rsts

    const __bf16* mg = mpk + (size_t)g * (13 * FDIM * DDIM);  // this motif's packed W

    v8f cacc[4], macc[4];
#pragma unroll
    for (int nt = 0; nt < 4; ++nt)
#pragma unroll
        for (int t = 0; t < 8; ++t) { cacc[nt][t] = 0.f; macc[nt][t] = 0.f; }

    for (int j = 0; j < 14; ++j) {
        const __bf16* A = rsts + (size_t)j * N_NODES * FDIM;
        if (j == i) {
            // mw accumulator: rsts[i] @ wa
#pragma unroll
            for (int s = 0; s < 4; ++s) {
                v16bf a = load_a_bf16(A, m0, lane, s * 32);
#pragma unroll
                for (int nt = 0; nt < 4; ++nt) {
                    v16bf b = *(const v16bf*)(wapk + ((size_t)((s * 4 + nt) * 32 + lane) << 4));
                    macc[nt] = __builtin_amdgcn_wmma_f32_16x16x32_bf16(
                        false, a, false, b, (short)0, macc[nt], false, false);
                }
            }
        } else {
            int p = (j < i) ? j : j - 1;    // position of block j in the concat
            const __bf16* B = mg + (size_t)p * (FDIM * DDIM); // 4 kt * 4 nt * 512
#pragma unroll
            for (int s = 0; s < 4; ++s) {
                v16bf a = load_a_bf16(A, m0, lane, s * 32);
#pragma unroll
                for (int nt = 0; nt < 4; ++nt) {
                    v16bf b = *(const v16bf*)(B + ((size_t)((s * 4 + nt) * 32 + lane) << 4));
                    cacc[nt] = __builtin_amdgcn_wmma_f32_16x16x32_bf16(
                        false, a, false, b, (short)0, cacc[nt], false, false);
                }
            }
        }
    }

    // epilogue: biases, row-wise dot across 64 columns, sigmoid, output
    int col0 = lane & 15;
    float cv[4][8], mv[4][8];
#pragma unroll
    for (int nt = 0; nt < 4; ++nt) {
        float cb = motif_b[g * DDIM + nt * 16 + col0];
        float ab = ba[nt * 16 + col0];
#pragma unroll
        for (int r = 0; r < 8; ++r) {
            cv[nt][r] = cacc[nt][r] + cb;
            mv[nt][r] = macc[nt][r] + ab;
        }
    }
    float part[8];
#pragma unroll
    for (int r = 0; r < 8; ++r) {
        float s = 0.f;
#pragma unroll
        for (int nt = 0; nt < 4; ++nt) s += mv[nt][r] * cv[nt][r];
        part[r] = s;
    }
    // reduce across the 16 lanes of each half-wave (rows live per-half: masks 1,2,4,8)
#pragma unroll
    for (int r = 0; r < 8; ++r) {
        part[r] += __shfl_xor(part[r], 1, 32);
        part[r] += __shfl_xor(part[r], 2, 32);
        part[r] += __shfl_xor(part[r], 4, 32);
        part[r] += __shfl_xor(part[r], 8, 32);
    }
    int row_lo = m0 + ((lane >> 4) << 3);
#pragma unroll
    for (int r = 0; r < 8; ++r) {
        int row = row_lo + r;
        if (row >= N_NODES) continue;
        float att = 1.f / (1.f + __expf(-part[r]));
        float* op = out + (size_t)row * OUTC + g * DDIM;
#pragma unroll
        for (int nt = 0; nt < 4; ++nt)
            op[nt * 16 + col0] = att * (mv[nt][r] - cv[nt][r]);
    }
}

// ---------------------------------------------------------------------------
extern "C" void kernel_launch(void* const* d_in, const int* in_sizes, int n_in,
                              void* d_out, int out_size, void* d_ws, size_t ws_size,
                              hipStream_t stream) {
    const float*     x       = (const float*)d_in[0];
    const long long* ei      = (const long long*)d_in[1];
    const float*     ew      = (const float*)d_in[2];
    const float*     weight  = (const float*)d_in[3];
    const float*     root    = (const float*)d_in[4];
    const float*     bias    = (const float*)d_in[5];
    const float*     wa      = (const float*)d_in[6];
    const float*     ba      = (const float*)d_in[7];
    const float*     motif_w = (const float*)d_in[8];
    const float*     motif_b = (const float*)d_in[9];
    float* out = (float*)d_out;

    __bf16* rsts = (__bf16*)d_ws;                                  // 14*N*128
    float*  agg  = (float*)(rsts + (size_t)14 * N_NODES * FDIM);   // N*128
    __bf16* wpk  = (__bf16*)(agg + (size_t)N_NODES * FDIM);        // 128*128
    __bf16* rpk  = wpk + 128 * 128;                                 // 128*128
    __bf16* wapk = rpk + 128 * 128;                                 // 128*64
    __bf16* mpk  = wapk + 128 * 64;                                 // 13*1664*64

    // 1) zero the atomic accumulators (rsts bf16 + agg fp32); re-done each call
    long long n16 = ((long long)14 * N_NODES * FDIM * 2 + (long long)N_NODES * FDIM * 4) / 16;
    zero_kernel<<<(unsigned)((n16 + 255) / 256), 256, 0, stream>>>((uint4*)d_ws, n16);

    // 2) pack all static weights into bf16 WMMA-B fragment layout
    const int packTot = 128 * 128 * 2 + 128 * 64 + 13 * FDIM * DDIM * 13 / 13 * 13; // see below
    (void)packTot;
    int packTotal = 128 * 128 + 128 * 128 + 128 * 64 + 13 * (13 * FDIM * DDIM);
    pack_kernel<<<(packTotal + 255) / 256, 256, 0, stream>>>(weight, root, wa, motif_w,
                                                             wpk, rpk, wapk, mpk);

    // 3) edge-set-0 propagate (fp32 atomics into agg)
    prop0_kernel<<<(N_EDGES * 32) / 256, 256, 0, stream>>>(x, ei, ew, agg);

    // 4) h = agg@weight + x@root + bias  ->  rsts[0] bf16
    h_gemm_kernel<<<(N_NODES + 127) / 128, 256, 0, stream>>>(agg, x, bias, wpk, rpk, rsts);

    // 5) 13 motif propagates: bf16 gather from h, packed-bf16 atomic scatter
    long long tot = 13LL * N_EDGES * 64;
    motif_prop_kernel<<<(unsigned)(tot / 256), 256, 0, stream>>>(ei, ew, rsts, rsts);

    // 6) fused motif GEMM + attention + output
    dim3 mgrid((N_NODES + 127) / 128, 13);
    motif_gemm_kernel<<<mgrid, 256, 0, stream>>>(rsts, wapk, mpk, ba, motif_b, out);
}
